// DbrxExperts_36627481100906
// MI455X (gfx1250) — matrix operand order repaired and checked by reading, not compiled, not run on hardware
//
#include <hip/hip_runtime.h>
#include <hip/hip_bf16.h>
#include <math.h>

// Problem constants (match reference): T = B*S tokens
#define TT 8192
#define FF 2048
#define HH 4096
#define EE 8
#define KK 2

#define LDSTRIDE 40   // shorts per LDS row (32 data + 8 pad, 16B aligned)

typedef __attribute__((ext_vector_type(16))) __bf16 v16bf;
typedef __attribute__((ext_vector_type(8)))  float  v8f;

// low 32 bits of a generic pointer into LDS == LDS byte address
#define LDS_ADDR(p) ((unsigned)(unsigned long long)(const void*)(p))

// CDNA5 async DMA: copy 16B/lane from global directly into LDS (ASYNCcnt)
__device__ __forceinline__ void async_b128(unsigned lds,
                                           unsigned long long gaddr) {
  asm volatile("global_load_async_to_lds_b128 %0, %1, off"
               :: "v"(lds), "v"(gaddr) : "memory");
}
__device__ __forceinline__ void wait_async0() {
  asm volatile("s_wait_asynccnt 0x0" ::: "memory");
}

__device__ __forceinline__ unsigned short f32_to_bf16(float f) {
  unsigned int u = __float_as_uint(f);
  u += 0x7FFFu + ((u >> 16) & 1u);          // round-to-nearest-even
  return (unsigned short)(u >> 16);
}

// A-matrix fragment (16x32 bf16): lane holds row (lane&15), K runs
// [koff, koff+8) and [koff+16, koff+24), koff = (lane>>4)*8. Two ds_read_b128.
__device__ __forceinline__ v16bf frag_ld_a(const unsigned short* base, int row,
                                           int koff) {
  union { uint4 u[2]; v16bf v; } f;
  f.u[0] = *(const uint4*)(base + row * LDSTRIDE + koff);
  f.u[1] = *(const uint4*)(base + row * LDSTRIDE + koff + 16);
  return f.v;
}

// B-matrix fragment (32x16 bf16): lane holds col (lane&15), K runs
// [kbase, kbase+8) and [kbase+16, kbase+24), kbase = (lane>>4)*8. LDS tile layout: [n][k].
__device__ __forceinline__ v16bf frag_ld_b(const unsigned short* base, int col,
                                           int kbase) {
  union { uint4 u[2]; v16bf v; } f;
  f.u[0] = *(const uint4*)(base + col * LDSTRIDE + kbase);
  f.u[1] = *(const uint4*)(base + col * LDSTRIDE + kbase + 16);
  return f.v;
}

// ---------------------------------------------------------------- utilities
__global__ void k_zero(unsigned int* __restrict__ p, long n) {
  long i = (long)blockIdx.x * blockDim.x + threadIdx.x;
  if (i < n) p[i] = 0u;
}

__global__ void k_scatter(const int* __restrict__ idx,
                          const float* __restrict__ w,
                          float* __restrict__ combine) {
  int i = blockIdx.x * blockDim.x + threadIdx.x;
  if (i >= TT * KK) return;
  int t = i / KK;
  int e = idx[i];
  atomicAdd(combine + t * EE + e, w[i]);
}

__global__ void k_build(const float* __restrict__ combine,
                        int* __restrict__ counts,
                        int* __restrict__ tok,
                        float* __restrict__ wgt) {
  int t = blockIdx.x * blockDim.x + threadIdx.x;
  if (t >= TT) return;
  for (int e = 0; e < EE; ++e) {
    float c = combine[t * EE + e];
    if (c != 0.f) {
      int p = atomicAdd(counts + e, 1);
      tok[e * TT + p] = t;
      wgt[e * TT + p] = c;
    }
  }
}

__global__ void k_cvt_bf16(const float* __restrict__ in,
                           unsigned short* __restrict__ out, long n4) {
  long i = (long)blockIdx.x * blockDim.x + threadIdx.x;
  if (i >= n4) return;
  float4 f = ((const float4*)in)[i];
  ushort4 o;
  o.x = f32_to_bf16(f.x);
  o.y = f32_to_bf16(f.y);
  o.z = f32_to_bf16(f.z);
  o.w = f32_to_bf16(f.w);
  ((ushort4*)out)[i] = o;
}

// src [E][F][H] fp32  ->  dst [E][H][F] bf16 (tiled transpose; used for w1,v1)
__global__ void k_tr(const float* __restrict__ src,
                     unsigned short* __restrict__ dst) {
  __shared__ unsigned short tile[32][33];
  const int e  = blockIdx.z;
  const int h0 = blockIdx.x * 32;
  const int f0 = blockIdx.y * 32;
  const int x  = threadIdx.x;
#pragma unroll
  for (int j = 0; j < 32; j += 8) {
    int f = f0 + threadIdx.y + j;
    tile[threadIdx.y + j][x] =
        f32_to_bf16(src[((size_t)e * FF + f) * HH + h0 + x]);
  }
  __syncthreads();
#pragma unroll
  for (int j = 0; j < 32; j += 8) {
    int h = h0 + threadIdx.y + j;
    dst[((size_t)e * HH + h) * FF + f0 + x] = tile[x][threadIdx.y + j];
  }
}

// -------------------------------------------- fused gate/up expert GEMM
// act[p, h] = silu(x[tok[p]]·w1e)[h] * (x[tok[p]]·v1e)[h], bf16
// Tiles staged via CDNA5 async global->LDS DMA; 8 waves, 32x32 region each.
__global__ __launch_bounds__(256) void k_gateup(
    const unsigned short* __restrict__ xb,
    const unsigned short* __restrict__ w1t,
    const unsigned short* __restrict__ v1t,
    const int* __restrict__ counts,
    const int* __restrict__ tok,
    unsigned short* __restrict__ act, int e) {
  __shared__ unsigned short As[2][128 * LDSTRIDE];  // 128 rows x 32 K
  __shared__ unsigned short Bw[2][64 * LDSTRIDE];   // [n][k]
  __shared__ unsigned short Bv[2][64 * LDSTRIDE];
  __shared__ int s_tok[128];

  const int cnt  = counts[e];
  const int row0 = blockIdx.x * 128;
  if (row0 >= cnt) return;
  const int hbase = blockIdx.y * 64;
  const unsigned short* w1e = w1t + (size_t)e * HH * FF;
  const unsigned short* v1e = v1t + (size_t)e * HH * FF;
  const int* toke = tok + e * TT;

  const int tid = threadIdx.x;
  if (tid < 128) {
    int p = row0 + tid;
    s_tok[tid] = (p < cnt) ? toke[p] : toke[0];  // clamp: in-bounds reads
  }
  __syncthreads();

  // async staging slots: A tile = 512 x 16B slots (2/thread),
  // B tiles (w1 + v1) = 512 x 16B slots (2/thread).
  // A slot s: row = s>>2, quarter = s&3 (8 shorts each)
  const int ar0 = tid >> 2,          aq0 = (tid & 3) * 8;
  const int ar1 = (tid + 256) >> 2,  aq1 = ((tid + 256) & 3) * 8;
  const int br  = tid >> 2,          bq  = (tid & 3) * 8;

  // running global addresses (advance 64B per K-step)
  unsigned long long a_g0 =
      (unsigned long long)(xb + (size_t)s_tok[ar0] * FF + aq0);
  unsigned long long a_g1 =
      (unsigned long long)(xb + (size_t)s_tok[ar1] * FF + aq1);
  unsigned long long b_g0 =
      (unsigned long long)(w1e + (size_t)(hbase + br) * FF + bq);
  unsigned long long b_g1 =
      (unsigned long long)(v1e + (size_t)(hbase + br) * FF + bq);
  const unsigned a_l0 = LDS_ADDR(&As[0][ar0 * LDSTRIDE + aq0]);
  const unsigned a_l1 = LDS_ADDR(&As[0][ar1 * LDSTRIDE + aq1]);
  const unsigned b_l0 = LDS_ADDR(&Bw[0][br * LDSTRIDE + bq]);
  const unsigned b_l1 = LDS_ADDR(&Bv[0][br * LDSTRIDE + bq]);

  auto issue = [&](int buf) {
    const unsigned ao = (unsigned)buf * (unsigned)sizeof(As[0]);
    const unsigned bo = (unsigned)buf * (unsigned)sizeof(Bw[0]);
    async_b128(a_l0 + ao, a_g0);
    async_b128(a_l1 + ao, a_g1);
    async_b128(b_l0 + bo, b_g0);
    async_b128(b_l1 + bo, b_g1);
    a_g0 += 64; a_g1 += 64; b_g0 += 64; b_g1 += 64;
  };

  const int lane  = tid & 31;
  const int wave  = tid >> 5;
  const int koff  = (lane >> 4) * 8;    // A fragment K offset
  const int kbase = (lane >> 4) * 8;    // B fragment K offset
  const int nlo   = lane & 15;
  const int m0    = (wave >> 1) * 32;   // 4 m-blocks x 2 n-blocks of 32x32
  const int n0    = (wave & 1) * 32;
  const int rA0   = m0 + (lane & 15);
  const int rA1   = rA0 + 16;

  v8f cg[2][2] = {};
  v8f cu[2][2] = {};

  issue(0);
#pragma unroll 1   // no unroll: keeps accumulators tied (no v_mov/v_nop churn)
  for (int kb = 0; kb < FF / 32; ++kb) {
    wait_async0();                     // own DMAs into buf kb&1 done
    __syncthreads();                   // everyone's done + visible
    if (kb + 1 < FF / 32) issue((kb + 1) & 1);  // overlap with WMMAs

    const unsigned short* AS = As[kb & 1];
    v16bf a0 = frag_ld_a(AS, rA0, koff);
    v16bf a1 = frag_ld_a(AS, rA1, koff);
#pragma unroll
    for (int nt = 0; nt < 2; ++nt) {
      const int col = n0 + nt * 16 + nlo;
      v16bf bw = frag_ld_b(Bw[kb & 1], col, kbase);
      v16bf bv = frag_ld_b(Bv[kb & 1], col, kbase);
      cg[0][nt] = __builtin_amdgcn_wmma_f32_16x16x32_bf16(
          false, a0, false, bw, (short)0, cg[0][nt], false, false);
      cg[1][nt] = __builtin_amdgcn_wmma_f32_16x16x32_bf16(
          false, a1, false, bw, (short)0, cg[1][nt], false, false);
      cu[0][nt] = __builtin_amdgcn_wmma_f32_16x16x32_bf16(
          false, a0, false, bv, (short)0, cu[0][nt], false, false);
      cu[1][nt] = __builtin_amdgcn_wmma_f32_16x16x32_bf16(
          false, a1, false, bv, (short)0, cu[1][nt], false, false);
    }
  }

#pragma unroll
  for (int mi = 0; mi < 2; ++mi) {
#pragma unroll
    for (int nt = 0; nt < 2; ++nt) {
      const int col = hbase + n0 + nt * 16 + nlo;
#pragma unroll
      for (int i = 0; i < 8; ++i) {
        int m = m0 + mi * 16 + ((lane >> 4) * 8) + i;  // lanes16-31 -> M=8+i
        int p = row0 + m;
        float g = cg[mi][nt][i];
        float u = cu[mi][nt][i];
        float s = g * __builtin_amdgcn_rcpf(1.0f + __expf(-g));  // fast silu
        act[(size_t)p * HH + col] = f32_to_bf16(s * u);
      }
    }
  }
}

// -------------------------------------------- down projection + scatter
// out[tok[p], f] += wgt[p] * (act[p]·w2e^T)[f]
// B = w2 native [E][F][H]: output col f is a K(=H)-contiguous row already.
__global__ __launch_bounds__(256) void k_down(
    const unsigned short* __restrict__ act,
    const unsigned short* __restrict__ w2b,
    const int* __restrict__ counts,
    const int* __restrict__ tok,
    const float* __restrict__ wgt,
    float* __restrict__ out, int e) {
  __shared__ unsigned short As[2][128 * LDSTRIDE];
  __shared__ unsigned short Bt[2][64 * LDSTRIDE];
  __shared__ int   s_tok[128];
  __shared__ float s_w[128];

  const int cnt  = counts[e];
  const int row0 = blockIdx.x * 128;
  if (row0 >= cnt) return;
  const int fbase = blockIdx.y * 64;
  const unsigned short* w2e = w2b + (size_t)e * FF * HH;
  const int*   toke = tok + e * TT;
  const float* wgte = wgt + e * TT;

  const int tid = threadIdx.x;
  if (tid < 128) {
    int p = row0 + tid;
    bool v = p < cnt;
    s_tok[tid] = v ? toke[p] : 0;
    s_w[tid]   = v ? wgte[p] : 0.f;
  }
  __syncthreads();

  // A tile: 512 slots (2/thread); B tile: 256 slots (1/thread)
  const int ar0 = tid >> 2,          aq0 = (tid & 3) * 8;
  const int ar1 = (tid + 256) >> 2,  aq1 = ((tid + 256) & 3) * 8;
  const int br  = tid >> 2,          bq  = (tid & 3) * 8;

  unsigned long long a_g0 =
      (unsigned long long)(act + (size_t)(row0 + ar0) * HH + aq0);
  unsigned long long a_g1 =
      (unsigned long long)(act + (size_t)(row0 + ar1) * HH + aq1);
  unsigned long long b_g0 =
      (unsigned long long)(w2e + (size_t)(fbase + br) * HH + bq);
  const unsigned a_l0 = LDS_ADDR(&As[0][ar0 * LDSTRIDE + aq0]);
  const unsigned a_l1 = LDS_ADDR(&As[0][ar1 * LDSTRIDE + aq1]);
  const unsigned b_l0 = LDS_ADDR(&Bt[0][br * LDSTRIDE + bq]);

  auto issue = [&](int buf) {
    const unsigned ao = (unsigned)buf * (unsigned)sizeof(As[0]);
    const unsigned bo = (unsigned)buf * (unsigned)sizeof(Bt[0]);
    async_b128(a_l0 + ao, a_g0);
    async_b128(a_l1 + ao, a_g1);
    async_b128(b_l0 + bo, b_g0);
    a_g0 += 64; a_g1 += 64; b_g0 += 64;
  };

  const int lane  = tid & 31;
  const int wave  = tid >> 5;
  const int koff  = (lane >> 4) * 8;
  const int kbase = (lane >> 4) * 8;
  const int nlo   = lane & 15;
  const int m0    = (wave >> 1) * 32;
  const int n0    = (wave & 1) * 32;
  const int rA0   = m0 + (lane & 15);
  const int rA1   = rA0 + 16;

  v8f c[2][2] = {};

  issue(0);
#pragma unroll 1
  for (int kb = 0; kb < HH / 32; ++kb) {
    wait_async0();
    __syncthreads();
    if (kb + 1 < HH / 32) issue((kb + 1) & 1);

    const unsigned short* AS = As[kb & 1];
    v16bf a0 = frag_ld_a(AS, rA0, koff);
    v16bf a1 = frag_ld_a(AS, rA1, koff);
#pragma unroll
    for (int nt = 0; nt < 2; ++nt) {
      const int col = n0 + nt * 16 + nlo;
      v16bf b = frag_ld_b(Bt[kb & 1], col, kbase);
      c[0][nt] = __builtin_amdgcn_wmma_f32_16x16x32_bf16(
          false, a0, false, b, (short)0, c[0][nt], false, false);
      c[1][nt] = __builtin_amdgcn_wmma_f32_16x16x32_bf16(
          false, a1, false, b, (short)0, c[1][nt], false, false);
    }
  }

#pragma unroll
  for (int mi = 0; mi < 2; ++mi) {
#pragma unroll
    for (int nt = 0; nt < 2; ++nt) {
      const int col = fbase + n0 + nt * 16 + nlo;
#pragma unroll
      for (int i = 0; i < 8; ++i) {
        int m = m0 + mi * 16 + ((lane >> 4) * 8) + i;
        int p = row0 + m;
        if (p < cnt) {
          int   t = s_tok[m];
          float w = s_w[m];
          float* o = out + (size_t)t * FF + col;
          *o += w * c[mi][nt][i];   // rows unique per expert; experts serialize
        }
      }
    }
  }
}

// ---------------------------------------------------------------- launcher
extern "C" void kernel_launch(void* const* d_in, const int* in_sizes, int n_in,
                              void* d_out, int out_size, void* d_ws,
                              size_t ws_size, hipStream_t stream) {
  const float* x   = (const float*)d_in[0];   // hidden_states [B,S,F]
  const float* tkw = (const float*)d_in[1];   // top_k_weights [T,K]
  const float* w1  = (const float*)d_in[2];   // [E,F,H]
  const float* v1  = (const float*)d_in[3];   // [E,F,H]
  const float* w2  = (const float*)d_in[4];   // [E,F,H]
  const int*   tki = (const int*)d_in[5];     // top_k_index [T,K]
  float* out = (float*)d_out;
  (void)in_sizes; (void)n_in; (void)out_size; (void)ws_size;

  char* ws = (char*)d_ws;
  size_t off = 0;
  auto carve = [&](size_t bytes) -> char* {
    char* p = ws + off;
    off += (bytes + 255) & ~(size_t)255;
    return p;
  };
  float* combine        = (float*)carve((size_t)TT * EE * 4);
  int*   counts         = (int*)carve(EE * 4);
  int*   tok            = (int*)carve((size_t)EE * TT * 4);
  float* wgt            = (float*)carve((size_t)EE * TT * 4);
  unsigned short* xb    = (unsigned short*)carve((size_t)TT * FF * 2);
  unsigned short* w1t   = (unsigned short*)carve((size_t)EE * HH * FF * 2);
  unsigned short* v1t   = (unsigned short*)carve((size_t)EE * HH * FF * 2);
  unsigned short* w2b   = (unsigned short*)carve((size_t)EE * FF * HH * 2);
  unsigned short* act   = (unsigned short*)carve((size_t)TT * HH * 2);

  const long nOut = (long)TT * FF;
  k_zero<<<(unsigned)((nOut + 255) / 256), 256, 0, stream>>>(
      (unsigned int*)out, nOut);
  k_zero<<<(TT * EE + 255) / 256, 256, 0, stream>>>(
      (unsigned int*)combine, (long)TT * EE);
  k_zero<<<1, 32, 0, stream>>>((unsigned int*)counts, (long)EE);

  k_scatter<<<(TT * KK + 255) / 256, 256, 0, stream>>>(tki, tkw, combine);
  k_build<<<(TT + 255) / 256, 256, 0, stream>>>(combine, counts, tok, wgt);

  const long n4x = (long)TT * FF / 4;
  k_cvt_bf16<<<(unsigned)((n4x + 255) / 256), 256, 0, stream>>>(x, xb, n4x);
  // w1, v1: transpose to [E][H][F] bf16 so GEMM B-tiles are K-contiguous
  k_tr<<<dim3(HH / 32, FF / 32, EE), dim3(32, 8), 0, stream>>>(w1, w1t);
  k_tr<<<dim3(HH / 32, FF / 32, EE), dim3(32, 8), 0, stream>>>(v1, v1t);
  // w2: native [E][F][H] layout is already what the down-GEMM B wants
  const long n4w = (long)EE * FF * HH / 4;
  k_cvt_bf16<<<(unsigned)((n4w + 255) / 256), 256, 0, stream>>>(w2, w2b, n4w);

  for (int e = 0; e < EE; ++e) {
    k_gateup<<<dim3(TT / 128, HH / 64), 256, 0, stream>>>(
        xb, w1t, v1t, counts, tok, act, e);
    k_down<<<dim3(TT / 128, FF / 64), 256, 0, stream>>>(
        act, w2b, counts, tok, wgt, out, e);
  }
}
